// NodeLevelAttention_51436528337337
// MI455X (gfx1250) — compile-verified, matrix operand order checked by reading, not run on hardware
//
#include <hip/hip_runtime.h>
#include <math.h>

#define N_NODES  50000
#define IN_DIM   128
#define HEADS    4
#define HEAD_DIM 32
#define TYPE_NUM 3
#define EDGES    800000
#define NEG_SLOPE 0.2f

typedef __attribute__((ext_vector_type(2)))  float    v2f;
typedef __attribute__((ext_vector_type(8)))  float    v8f;
typedef __attribute__((ext_vector_type(16))) _Float16 v16h;

#if defined(__has_builtin)
#if __has_builtin(__builtin_amdgcn_wmma_f32_16x16x4_f32)
#define USE_WMMA_F32 1
#endif
#endif

// ---------------------------------------------------------------------------
// Kernel 1: h[t] = x @ W[t]   ([N,128] x [128,128] -> [N,128]) via WMMA.
// Block = 256 threads = 8 wave32; block tile = 64 rows x 32 cols.
// Wave (wr, wc) computes a 16x16 tile, K swept 128 in steps of 4 (f32 WMMA).
// LDS: x tile [64][132] + W^T tile [32][132] (pad 4 to avoid bank conflicts).
// ---------------------------------------------------------------------------
__global__ __launch_bounds__(256) void gat_gemm_wmma(
    const float* __restrict__ x, const float* __restrict__ W,
    float* __restrict__ h)
{
    __shared__ float x_lds[64 * 132];
    __shared__ float w_lds[32 * 132];

    const int t   = blockIdx.z;
    const int m0  = blockIdx.x * 64;
    const int n0  = blockIdx.y * 32;
    const int tid = threadIdx.x;
    const float* Wt = W + (size_t)t * IN_DIM * IN_DIM;  // [k=128][n=128] (h*32+d flattened)

    // Stage x tile (64 rows x 128 K), float4 loads, zero-pad OOB rows.
    for (int i = tid; i < 64 * 32; i += 256) {
        const int row = i >> 5;
        const int k4  = (i & 31) << 2;
        float4 v = make_float4(0.f, 0.f, 0.f, 0.f);
        const int g = m0 + row;
        if (g < N_NODES) v = *(const float4*)(x + (size_t)g * IN_DIM + k4);
        *(float4*)(&x_lds[row * 132 + k4]) = v;
    }
    // Stage W tile transposed: w_lds[nLocal][k] for nLocal in [0,32).
    for (int i = tid; i < 128 * 32; i += 256) {
        const int k  = i >> 5;
        const int nl = i & 31;
        w_lds[nl * 132 + k] = Wt[(size_t)k * IN_DIM + (n0 + nl)];
    }
    __syncthreads();

    const int wave = tid >> 5;
    const int lane = tid & 31;
    const int wr   = wave & 3;    // row sub-tile 0..3
    const int wc   = wave >> 2;   // col sub-tile 0..1
    const int l15  = lane & 15;
    const int hi   = lane >> 4;   // lane half: 0 -> K{0,1}/M{0..7}, 1 -> K{2,3}/M{8..15}

    const float* arow = &x_lds[(wr * 16 + l15) * 132];  // A row (M = l15)
    const float* brow = &w_lds[(wc * 16 + l15) * 132];  // B col (N = l15), K-major

    v8f acc = {};
#if defined(USE_WMMA_F32)
    const int koff = hi * 2;   // f32 A/B: lanes 0-15 hold K+0..1, lanes 16-31 hold K+2..3
#pragma unroll
    for (int k = 0; k < IN_DIM; k += 4) {
        v2f a = *(const v2f*)(arow + k + koff);
        v2f b = *(const v2f*)(brow + k + koff);
        acc = __builtin_amdgcn_wmma_f32_16x16x4_f32(
            false, a, false, b, (short)0, acc, false, false);
    }
#else
    // Fallback: probe-confirmed f16 WMMA (16x16x32), convert in registers.
    const int koff8 = hi * 8;  // 16-bit A layout: halves 0-7 -> K+0..7, halves 8-15 -> K+16..23
#pragma unroll
    for (int k = 0; k < IN_DIM; k += 32) {
        v16h a, b;
#pragma unroll
        for (int i = 0; i < 16; ++i) {
            const int kk = k + koff8 + (i < 8 ? i : i + 8);
            a[i] = (_Float16)arow[kk];
            b[i] = (_Float16)brow[kk];
        }
        acc = __builtin_amdgcn_wmma_f32_16x16x32_f16(
            false, a, false, b, (short)0, acc, false, false);
    }
#endif

    // C/D layout: VGPR v -> M = v + 8*hi ; N = l15.
    const int grow_base = m0 + wr * 16 + hi * 8;
    const int gcol      = n0 + wc * 16 + l15;
    float* hb = h + (size_t)t * N_NODES * IN_DIM;
#pragma unroll
    for (int v = 0; v < 8; ++v) {
        const int grow = grow_base + v;
        if (grow < N_NODES) hb[(size_t)grow * IN_DIM + gcol] = acc[v];
    }
}

// ---------------------------------------------------------------------------
// Kernel 2: per (t, node, head) attention logits  e_src / e_dst.
// ---------------------------------------------------------------------------
__global__ void gat_esed(const float* __restrict__ h,
                         const float* __restrict__ a_src,
                         const float* __restrict__ a_dst,
                         float* __restrict__ es, float* __restrict__ ed)
{
    const int idx = blockIdx.x * blockDim.x + threadIdx.x; // t*N*H + node*H + head
    if (idx >= TYPE_NUM * N_NODES * HEADS) return;
    const int head = idx & (HEADS - 1);
    const int node = (idx >> 2) % N_NODES;
    const int t    = idx / (N_NODES * HEADS);

    const float* hp = h + ((size_t)(t * N_NODES + node)) * IN_DIM + head * HEAD_DIM;
    const float* as = a_src + t * IN_DIM + head * HEAD_DIM;
    const float* ad = a_dst + t * IN_DIM + head * HEAD_DIM;
    float s = 0.f, d = 0.f;
#pragma unroll
    for (int i = 0; i < HEAD_DIM; ++i) { const float v = hp[i]; s += v * as[i]; d += v * ad[i]; }
    es[idx] = s;
    ed[idx] = d;
}

__global__ void gat_init_minf(float* __restrict__ m, int n)
{
    const int idx = blockIdx.x * blockDim.x + threadIdx.x;
    if (idx < n) m[idx] = -INFINITY;
}

__device__ __forceinline__ float lrelu(float v) { return v > 0.f ? v : NEG_SLOPE * v; }

// Float atomic-max via IEEE ordering trick (no CAS loop).
__device__ __forceinline__ void atomicMaxFloat(float* addr, float v)
{
    if (v >= 0.f) atomicMax((int*)addr, __float_as_int(v));
    else          atomicMin((unsigned int*)addr, (unsigned int)__float_as_int(v));
}

// ---------------------------------------------------------------------------
// Kernel 3: segment max of edge logits into m[dst].
// ---------------------------------------------------------------------------
__global__ void gat_edge_max(const int* __restrict__ edges,
                             const float* __restrict__ es, const float* __restrict__ ed,
                             float* __restrict__ m)
{
    const int idx = blockIdx.x * blockDim.x + threadIdx.x;
    if (idx >= TYPE_NUM * EDGES) return;
    const int t = idx / EDGES;
    const int e = idx - t * EDGES;
    const int* eb = edges + (size_t)t * 2 * EDGES;
    const int s = eb[e], d = eb[EDGES + e];
    const int sb = (t * N_NODES + s) * HEADS;
    const int db = (t * N_NODES + d) * HEADS;
#pragma unroll
    for (int hh = 0; hh < HEADS; ++hh)
        atomicMaxFloat(&m[db + hh], lrelu(es[sb + hh] + ed[db + hh]));
}

// ---------------------------------------------------------------------------
// Kernel 4: wave-per-edge fused softmax-weight + denominator + aggregation.
// Lanes 0..3 compute the 4 head weights, accumulate them into den[dst]
// (the softmax denominator is the segment-sum of the same weights), then
// wave32 shuffles broadcast the weights and all 32 lanes scatter 4 dims
// each (coalesced 128-float gather + scatter-add).
// ---------------------------------------------------------------------------
__global__ __launch_bounds__(256) void gat_edge_agg(
    const int* __restrict__ edges,
    const float* __restrict__ es, const float* __restrict__ ed,
    const float* __restrict__ m, const float* __restrict__ h,
    float* __restrict__ den, float* __restrict__ agg)
{
    const int t    = blockIdx.z;
    const int e    = blockIdx.x * 8 + (threadIdx.x >> 5);
    const int lane = threadIdx.x & 31;
    if (e >= EDGES) return;

    const int* eb = edges + (size_t)t * 2 * EDGES;
    const int s = eb[e], d = eb[EDGES + e];
    const int sb = (t * N_NODES + s) * HEADS;
    const int db = (t * N_NODES + d) * HEADS;

    float wv = 0.f;
    if (lane < HEADS) {
        wv = expf(lrelu(es[sb + lane] + ed[db + lane]) - m[db + lane]);
        atomicAdd(&den[db + lane], wv);   // fused denominator accumulation
    }
    const float w0 = __shfl(wv, 0, 32);
    const float w1 = __shfl(wv, 1, 32);
    const float w2 = __shfl(wv, 2, 32);
    const float w3 = __shfl(wv, 3, 32);

    const float* hs = h   + ((size_t)(t * N_NODES + s)) * IN_DIM;
    float*       ag = agg + ((size_t)(t * N_NODES + d)) * IN_DIM;
    atomicAdd(&ag[lane +  0], hs[lane +  0] * w0);
    atomicAdd(&ag[lane + 32], hs[lane + 32] * w1);
    atomicAdd(&ag[lane + 64], hs[lane + 64] * w2);
    atomicAdd(&ag[lane + 96], hs[lane + 96] * w3);
}

// ---------------------------------------------------------------------------
// Kernel 5: out = elu(agg / (denom + 1e-9)); duplicate into the gcn half.
// agg aliases out's first half (read-before-write per thread, no hazard).
// ---------------------------------------------------------------------------
__global__ void gat_final(const float* __restrict__ agg,
                          const float* __restrict__ den,
                          float* __restrict__ out)
{
    const int idx = blockIdx.x * blockDim.x + threadIdx.x;
    const int total = TYPE_NUM * N_NODES * IN_DIM;
    if (idx >= total) return;
    const int dim  = idx & (IN_DIM - 1);
    const int node = (idx >> 7) % N_NODES;
    const int t    = idx / (N_NODES * IN_DIM);
    const int head = dim >> 5;

    const float v = agg[idx] / (den[(t * N_NODES + node) * HEADS + head] + 1e-9f);
    const float r = v > 0.f ? v : (expf(v) - 1.f);
    out[idx] = r;
    out[(size_t)total + idx] = r;
}

// ---------------------------------------------------------------------------
extern "C" void kernel_launch(void* const* d_in, const int* in_sizes, int n_in,
                              void* d_out, int out_size, void* d_ws, size_t ws_size,
                              hipStream_t stream)
{
    const float* x     = (const float*)d_in[0];  // [N,128]
    const int*   edges = (const int*)  d_in[1];  // [3,2,E]
    const float* W     = (const float*)d_in[2];  // [3,128,4,32]
    const float* a_src = (const float*)d_in[3];  // [3,4,32]
    const float* a_dst = (const float*)d_in[4];  // [3,4,32]
    float* out = (float*)d_out;                  // [2,3,N,128]

    const size_t hN = (size_t)TYPE_NUM * N_NODES * IN_DIM;  // 19.2M floats
    const size_t eN = (size_t)TYPE_NUM * N_NODES * HEADS;   // 600k floats

    float* h   = (float*)d_ws;
    float* es  = h  + hN;
    float* ed  = es + eN;
    float* m   = ed + eN;
    float* den = m  + eN;
    float* agg = out;  // accumulate directly into the gat half of the output

    hipMemsetAsync(agg, 0, hN * sizeof(float), stream);
    hipMemsetAsync(den, 0, eN * sizeof(float), stream);
    gat_init_minf<<<(int)((eN + 255) / 256), 256, 0, stream>>>(m, (int)eN);

    gat_gemm_wmma<<<dim3((N_NODES + 63) / 64, IN_DIM / 32, TYPE_NUM), 256, 0, stream>>>(x, W, h);

    gat_esed<<<(int)((eN + 255) / 256), 256, 0, stream>>>(h, a_src, a_dst, es, ed);

    const int teBlocks = (TYPE_NUM * EDGES + 255) / 256;
    gat_edge_max<<<teBlocks, 256, 0, stream>>>(edges, es, ed, m);

    gat_edge_agg<<<dim3((EDGES + 7) / 8, 1, TYPE_NUM), 256, 0, stream>>>(
        edges, es, ed, m, h, den, agg);

    gat_final<<<(int)((hN + 255) / 256), 256, 0, stream>>>(agg, den, out);
}
